// LSTM_Cat_CNN_45363444580400
// MI455X (gfx1250) — compile-verified, matrix operand order checked by reading, not run on hardware
//
#include <hip/hip_runtime.h>
#include <hip/hip_bf16.h>

typedef __bf16 bf16_t;
typedef __attribute__((ext_vector_type(16))) __bf16 v16bf;
typedef __attribute__((ext_vector_type(8)))  float  v8f;
typedef int v4i __attribute__((vector_size(16)));   // matches builtin's V4i

#define T_TOT   20480
#define D_IN    868
#define D_PAD   896      // 868 padded to multiple of 32
#define H_DIM   256
#define G4      1024     // 4*H
#define N_DOCS  128
#define LMAX    200
#define NROWS   (N_DOCS * LMAX)   // 25600
#define NKF     128
#define OUT_COLS 896     // 384 cnn + 512 lstm
#define CONV_ROWS 1920   // (3+5+7)*128 packed weight rows
#define NBLK_LSTM 16

// ---- optional CDNA5 async global->LDS copy path ---------------------------
#if defined(__has_builtin)
# if __has_builtin(__builtin_amdgcn_global_load_async_to_lds_b128) && \
     __has_builtin(__builtin_amdgcn_s_wait_asynccnt)
#  define USE_ASYNC_COPY 1
# endif
#endif
#ifndef USE_ASYNC_COPY
# define USE_ASYNC_COPY 0
#endif

// copy 16 bytes (8 bf16) global -> LDS
__device__ __forceinline__ void copy16_g2l(const bf16_t* g, bf16_t* l) {
#if USE_ASYNC_COPY
    __builtin_amdgcn_global_load_async_to_lds_b128(
        (__attribute__((address_space(1))) v4i*)g,
        (__attribute__((address_space(3))) v4i*)l, 0, 0);
#else
    *reinterpret_cast<uint4*>(l) = *reinterpret_cast<const uint4*>(g);
#endif
}
// wait for this wave's async copies, then block barrier
__device__ __forceinline__ void stage_join() {
#if USE_ASYNC_COPY
    __builtin_amdgcn_s_wait_asynccnt(0);
#endif
    __syncthreads();
}

// ---- load two contiguous 16-byte chunks into a 16-half fragment -----------
__device__ __forceinline__ v16bf load2x8(const bf16_t* p0, const bf16_t* p1) {
    union { uint4 u[2]; v16bf v; } r;
    r.u[0] = *reinterpret_cast<const uint4*>(p0);
    r.u[1] = *reinterpret_cast<const uint4*>(p1);
    return r.v;
}
__device__ __forceinline__ void zero_acc(v8f& a) {
#pragma unroll
    for (int i = 0; i < 8; ++i) a[i] = 0.0f;
}
__device__ __forceinline__ float sigm(float x) { return 1.0f / (1.0f + __expf(-x)); }

// ===========================================================================
// meta: offsets (exclusive prefix of doc_lens), doc_id[t], pos[t]
// ===========================================================================
__global__ void k_meta(const int* __restrict__ doc_lens, int* offs,
                       int* doc_id, int* pos) {
    __shared__ int so[N_DOCS + 1];
    for (int t = threadIdx.x; t < T_TOT; t += blockDim.x) { doc_id[t] = 0; pos[t] = 0; }
    if (threadIdx.x == 0) {
        so[0] = 0;
        for (int i = 0; i < N_DOCS; ++i) so[i + 1] = so[i] + doc_lens[i];
    }
    __syncthreads();
    if (threadIdx.x <= N_DOCS) offs[threadIdx.x] = so[threadIdx.x];
    int d = threadIdx.x;
    if (d < N_DOCS) {
        int o = so[d], len = so[d + 1] - o;
        for (int tt = 0; tt < len; ++tt) {
            int t = o + tt;
            if (t < T_TOT) { doc_id[t] = d; pos[t] = tt; }
        }
    }
}

// ===========================================================================
// packing kernels (fp32 -> bf16, pad K to D_PAD)
// ===========================================================================
__global__ void k_pack_xpe(const float* __restrict__ x, const float* __restrict__ pos_table,
                           const int* __restrict__ pos, bf16_t* __restrict__ xpe) {
    long idx = (long)blockIdx.x * blockDim.x + threadIdx.x;
    if (idx >= (long)T_TOT * D_PAD) return;
    int t = (int)(idx / D_PAD), c = (int)(idx % D_PAD);
    float v = 0.0f;
    if (c < D_IN) {
        int p = pos[t]; if (p >= LMAX) p = LMAX - 1;
        v = x[(long)t * D_IN + c] + pos_table[(long)p * D_IN + c];
    }
    xpe[idx] = (bf16_t)v;
}

__global__ void k_pack_xd(const float* __restrict__ x, const int* __restrict__ doc_lens,
                          const int* __restrict__ offs,
                          bf16_t* __restrict__ xd, bf16_t* __restrict__ xr) {
    long idx = (long)blockIdx.x * blockDim.x + threadIdx.x;
    if (idx >= (long)NROWS * D_PAD) return;
    int row = (int)(idx / D_PAD), c = (int)(idx % D_PAD);
    int doc = row / LMAX, tt = row % LMAX;
    int len = doc_lens[doc], o = offs[doc];
    float vf = 0.0f, vr = 0.0f;
    if (c < D_IN && tt < len) {
        vf = x[(long)(o + tt) * D_IN + c];
        vr = x[(long)(o + len - 1 - tt) * D_IN + c];
    }
    xd[idx] = (bf16_t)vf;
    xr[idx] = (bf16_t)vr;
}

__global__ void k_pack_wih(const float* __restrict__ wf, const float* __restrict__ wb,
                           bf16_t* __restrict__ out) {
    long idx = (long)blockIdx.x * blockDim.x + threadIdx.x;
    if (idx >= 2L * G4 * D_PAD) return;
    int dir = (int)(idx / (G4 * D_PAD));
    int rem = (int)(idx % (G4 * D_PAD));
    int n = rem / D_PAD, k = rem % D_PAD;
    const float* src = dir ? wb : wf;
    out[idx] = (bf16_t)((k < D_IN) ? src[(long)n * D_IN + k] : 0.0f);
}

__global__ void k_pack_whh(const float* __restrict__ wf, const float* __restrict__ wb,
                           bf16_t* __restrict__ out) {
    long idx = (long)blockIdx.x * blockDim.x + threadIdx.x;
    if (idx >= 2L * G4 * H_DIM) return;
    int dir = (int)(idx / (G4 * H_DIM));
    int rem = (int)(idx % (G4 * H_DIM));
    out[idx] = (bf16_t)((dir ? wb : wf)[rem]);
}

__global__ void k_pack_conv(const float* __restrict__ w0, const float* __restrict__ w1,
                            const float* __restrict__ w2, bf16_t* __restrict__ out) {
    long idx = (long)blockIdx.x * blockDim.x + threadIdx.x;
    if (idx >= (long)CONV_ROWS * D_PAD) return;
    int row = (int)(idx / D_PAD), k = (int)(idx % D_PAD);
    const float* src; int taps, lrow;
    if (row < 3 * NKF)        { src = w0; taps = 3; lrow = row; }
    else if (row < 8 * NKF)   { src = w1; taps = 5; lrow = row - 3 * NKF; }
    else                      { src = w2; taps = 7; lrow = row - 8 * NKF; }
    int tap = lrow / NKF, n = lrow % NKF;
    out[idx] = (bf16_t)((k < D_IN) ? src[((long)n * D_IN + k) * taps + tap] : 0.0f);
}

__global__ void k_pack_bias(const float* bihf, const float* bhhf,
                            const float* bihb, const float* bhhb, float* out) {
    int idx = blockIdx.x * blockDim.x + threadIdx.x;
    if (idx >= 2 * G4) return;
    int dir = idx / G4, c = idx % G4;
    out[idx] = dir ? (bihb[c] + bhhb[c]) : (bihf[c] + bhhf[c]);
}

__global__ void k_zero(float* p, long n) {
    long idx = (long)blockIdx.x * blockDim.x + threadIdx.x;
    if (idx < n) p[idx] = 0.0f;
}

// ===========================================================================
// Gate-from-input GEMM:  G[dir] = xd/xr [25600, 896] @ Wih[dir].T + bias
// block = 128 thr (4 waves); tile 128(M) x 64(N); A+B staged via LDS.
// Each wave: 2 A-frags x 4 B-frags -> 8 wmma per K-step.
// ===========================================================================
__global__ void k_gemm_gates(const bf16_t* __restrict__ xd, const bf16_t* __restrict__ xr,
                             const bf16_t* __restrict__ wih, const float* __restrict__ gbias,
                             float* __restrict__ G) {
    const int dir = blockIdx.z;
    const bf16_t* A = dir ? xr : xd;
    const bf16_t* B = wih + (long)dir * G4 * D_PAD;
    float* Gout = G + (long)dir * NROWS * G4;
    const int mb = blockIdx.y * 128;
    const int nb = blockIdx.x * 64;
    const int wave = threadIdx.x >> 5, lane = threadIdx.x & 31;
    const int l = lane & 15, hi = lane >> 4;

    __shared__ __align__(16) bf16_t As[128 * 32];
    __shared__ __align__(16) bf16_t Bs[64 * 32];
    v8f acc[2][4];
#pragma unroll
    for (int mi = 0; mi < 2; ++mi)
#pragma unroll
        for (int nt = 0; nt < 4; ++nt) zero_acc(acc[mi][nt]);

    for (int kt = 0; kt < D_PAD / 32; ++kt) {
        __syncthreads();
        {   // stage A tile 128x32: each thread one row (4 x 16B)
            int row = threadIdx.x;
            const bf16_t* src = A + (long)(mb + row) * D_PAD + kt * 32;
#pragma unroll
            for (int ch = 0; ch < 4; ++ch)
                copy16_g2l(src + ch * 8, As + row * 32 + ch * 8);
            if (kt + 1 < D_PAD / 32)
                __builtin_prefetch(src + 32, 0, 1);
        }
        {   // stage B tile 64x32: two threads per row (2 x 16B each)
            int row = threadIdx.x >> 1, ch = threadIdx.x & 1;
            const bf16_t* src = B + (long)(nb + row) * D_PAD + kt * 32 + ch * 16;
            copy16_g2l(src, Bs + row * 32 + ch * 16);
            copy16_g2l(src + 8, Bs + row * 32 + ch * 16 + 8);
        }
        stage_join();
#pragma unroll
        for (int mi = 0; mi < 2; ++mi) {
            const bf16_t* ar = As + (wave * 32 + mi * 16 + l) * 32;
            v16bf af = load2x8(ar + hi * 8, ar + 16 + hi * 8);
#pragma unroll
            for (int nt = 0; nt < 4; ++nt) {
                const bf16_t* br = Bs + (nt * 16 + l) * 32;
                v16bf bfr = load2x8(br + hi * 8, br + 16 + hi * 8);
                acc[mi][nt] = __builtin_amdgcn_wmma_f32_16x16x32_bf16(
                    false, af, false, bfr, (short)0, acc[mi][nt], false, false);
            }
        }
    }
    const int n0 = lane & 15, moff = (lane >> 4) * 8;
#pragma unroll
    for (int mi = 0; mi < 2; ++mi)
#pragma unroll
        for (int nt = 0; nt < 4; ++nt)
#pragma unroll
            for (int r = 0; r < 8; ++r) {
                int row = mb + wave * 32 + mi * 16 + moff + r;
                int col = nb + nt * 16 + n0;
                Gout[(long)row * G4 + col] = acc[mi][nt][r] + gbias[dir * G4 + col];
            }
}

// ===========================================================================
// CNN branch as GEMM over taps: out[:, ks*128 .. +127]
// block = 128 thr; tile 64(M=T) x 128(N=filters); A+B staged via LDS.
// ===========================================================================
__global__ void k_conv(const bf16_t* __restrict__ xpe, const bf16_t* __restrict__ convw,
                       const float* __restrict__ b0, const float* __restrict__ b1,
                       const float* __restrict__ b2, float* __restrict__ out) {
    const int ks = blockIdx.y;                 // 0,1,2 -> k = 3,5,7
    const int taps = 3 + 2 * ks, half = ks + 1;
    const int wrow0 = (ks == 0) ? 0 : (ks == 1) ? 3 * NKF : 8 * NKF;
    const int mb = blockIdx.x * 64;
    const int wave = threadIdx.x >> 5, lane = threadIdx.x & 31;
    const int l = lane & 15, hi = lane >> 4;

    __shared__ __align__(16) bf16_t As[64 * 32];
    __shared__ __align__(16) bf16_t Bs[128 * 32];
    v8f acc[8];
#pragma unroll
    for (int i = 0; i < 8; ++i) zero_acc(acc[i]);

    for (int tap = 0; tap < taps; ++tap) {
        int shift = tap - half;
        for (int kt = 0; kt < D_PAD / 32; ++kt) {
            __syncthreads();
            {   // stage shifted A tile (zero outside [0,T)): 2 thr/row, 2x16B each
                int row = threadIdx.x >> 1, ch = threadIdx.x & 1;
                int trow = mb + row + shift;
                bf16_t* dst = As + row * 32 + ch * 16;
                if (trow >= 0 && trow < T_TOT) {
                    const bf16_t* src = xpe + (long)trow * D_PAD + kt * 32 + ch * 16;
                    copy16_g2l(src, dst);
                    copy16_g2l(src + 8, dst + 8);
                } else {
                    uint4 z; z.x = 0; z.y = 0; z.z = 0; z.w = 0;
                    reinterpret_cast<uint4*>(dst)[0] = z;
                    reinterpret_cast<uint4*>(dst)[1] = z;
                }
            }
            {   // stage B tile 128x32: one thread per row (4 x 16B)
                int row = threadIdx.x;
                const bf16_t* src = convw + (long)(wrow0 + tap * NKF + row) * D_PAD + kt * 32;
#pragma unroll
                for (int ch = 0; ch < 4; ++ch)
                    copy16_g2l(src + ch * 8, Bs + row * 32 + ch * 8);
            }
            stage_join();
            const bf16_t* ar = As + (wave * 16 + l) * 32;
            v16bf af = load2x8(ar + hi * 8, ar + 16 + hi * 8);
#pragma unroll
            for (int nt = 0; nt < 8; ++nt) {
                const bf16_t* br = Bs + (nt * 16 + l) * 32;
                v16bf bfr = load2x8(br + hi * 8, br + 16 + hi * 8);
                acc[nt] = __builtin_amdgcn_wmma_f32_16x16x32_bf16(
                    false, af, false, bfr, (short)0, acc[nt], false, false);
            }
        }
    }
    const float* bptr = (ks == 0) ? b0 : (ks == 1) ? b1 : b2;
    const int n0 = lane & 15, moff = (lane >> 4) * 8;
#pragma unroll
    for (int nt = 0; nt < 8; ++nt)
#pragma unroll
        for (int r = 0; r < 8; ++r) {
            int row = mb + wave * 16 + moff + r;
            int n = nt * 16 + n0;
            float v = acc[nt][r] + bptr[n];
            v = (v >= 0.0f) ? v : 0.01f * v;   // leaky relu
            out[(long)row * OUT_COLS + ks * NKF + n] = v;
        }
}

// ===========================================================================
// Persistent bidirectional LSTM: 16 co-resident blocks run all LMAX steps,
// separated by a device-scope atomic barrier. Whh slab (256x256) lives in
// LDS for the whole kernel; h_prev restaged (f32->bf16) each step.
// block b: hc=(b&3)*64, docs ((b>>2)&1)*64.., dir=b>>3
// ===========================================================================
__global__ void k_lstm_persist(const float* __restrict__ G, const bf16_t* __restrict__ whh,
                               float* __restrict__ hA, float* __restrict__ hB,
                               float* __restrict__ cstate, float* __restrict__ hout,
                               int* __restrict__ bar) {
    const int blk = blockIdx.x;
    const int hc = (blk & 3) * 64;
    const int dbase = ((blk >> 2) & 1) * 64;
    const int dir = blk >> 3;
    const float* Gd = G + (long)dir * NROWS * G4;
    const bf16_t* W = whh + (long)dir * G4 * H_DIM;
    float* cs = cstate + (long)dir * N_DOCS * H_DIM;
    float* ho = hout + (long)dir * N_DOCS * LMAX * H_DIM;

    extern __shared__ __align__(16) char smem[];
    bf16_t* Ws = (bf16_t*)smem;                          // 256 x 256 bf16 = 128 KB
    bf16_t* Hs = (bf16_t*)(smem + 256 * H_DIM * 2);      // 64 x 256 bf16  =  32 KB

    // stage the block's Whh slab once: local row = g*64 + j  (col j of this block)
    for (int e = threadIdx.x; e < 256 * 32; e += blockDim.x) {
        int rrow = e >> 5, ch = e & 31;                  // 32 chunks of 8 halves per row
        int g = rrow >> 6, j = rrow & 63;
        copy16_g2l(W + (long)(g * 256 + hc + j) * H_DIM + ch * 8,
                   Ws + rrow * H_DIM + ch * 8);
    }
    stage_join();

    const int wave = threadIdx.x >> 5, lane = threadIdx.x & 31;
    const int l = lane & 15, hi = lane >> 4;
    const int n0 = lane & 15, moff = (lane >> 4) * 8;

    for (int t = 0; t < LMAX; ++t) {
        const float* hp = ((t & 1) ? hB : hA) + (long)dir * N_DOCS * H_DIM;
        float*       hn = ((t & 1) ? hA : hB) + (long)dir * N_DOCS * H_DIM;

        // stage h_prev (f32 -> bf16), vectorized float4
        for (int e = threadIdx.x; e < 64 * (H_DIM / 4); e += blockDim.x) {
            int rr = e >> 6, c4 = (e & 63) * 4;
            const float4 v = *reinterpret_cast<const float4*>(&hp[(dbase + rr) * H_DIM + c4]);
            bf16_t* d = Hs + rr * H_DIM + c4;
            d[0] = (bf16_t)v.x; d[1] = (bf16_t)v.y; d[2] = (bf16_t)v.z; d[3] = (bf16_t)v.w;
        }
        __syncthreads();

        v8f acc[4][4];
#pragma unroll
        for (int g = 0; g < 4; ++g)
#pragma unroll
            for (int nt = 0; nt < 4; ++nt) zero_acc(acc[g][nt]);

        for (int kt = 0; kt < H_DIM / 32; ++kt) {
            const bf16_t* ar = Hs + (wave * 16 + l) * H_DIM + kt * 32;
            v16bf af = load2x8(ar + hi * 8, ar + 16 + hi * 8);
#pragma unroll
            for (int g = 0; g < 4; ++g)
#pragma unroll
                for (int nt = 0; nt < 4; ++nt) {
                    const bf16_t* br = Ws + (g * 64 + nt * 16 + l) * H_DIM + kt * 32;
                    v16bf bfr = load2x8(br + hi * 8, br + 16 + hi * 8);
                    acc[g][nt] = __builtin_amdgcn_wmma_f32_16x16x32_bf16(
                        false, af, false, bfr, (short)0, acc[g][nt], false, false);
                }
        }

#pragma unroll
        for (int nt = 0; nt < 4; ++nt) {
#pragma unroll
            for (int r = 0; r < 8; ++r) {
                int doc = dbase + wave * 16 + moff + r;
                int col = hc + nt * 16 + n0;
                const float* gr = Gd + ((long)doc * LMAX + t) * G4;
                float gi = acc[0][nt][r] + gr[0 * 256 + col];
                float gf = acc[1][nt][r] + gr[1 * 256 + col];
                float gg = acc[2][nt][r] + gr[2 * 256 + col];
                float go = acc[3][nt][r] + gr[3 * 256 + col];
                int ci = doc * H_DIM + col;
                float c = sigm(gf) * cs[ci] + sigm(gi) * tanhf(gg);
                cs[ci] = c;
                float h = sigm(go) * tanhf(c);
                hn[ci] = h;
                ho[((long)doc * LMAX + t) * H_DIM + col] = h;
            }
        }

        // device-scope barrier across the 16 blocks
        __syncthreads();
        if (threadIdx.x == 0) {
            __threadfence();
            __hip_atomic_fetch_add(bar, 1, __ATOMIC_RELEASE, __HIP_MEMORY_SCOPE_AGENT);
            const int target = NBLK_LSTM * (t + 1);
            while (__hip_atomic_load(bar, __ATOMIC_ACQUIRE, __HIP_MEMORY_SCOPE_AGENT) < target)
                __builtin_amdgcn_s_sleep(1);
            __threadfence();
        }
        __syncthreads();
    }
}

// ===========================================================================
// Final gather of LSTM outputs into d_out cols 384..895
// ===========================================================================
__global__ void k_gather(const float* __restrict__ hout, const int* __restrict__ doc_id,
                         const int* __restrict__ pos, const int* __restrict__ doc_lens,
                         float* __restrict__ out) {
    long idx = (long)blockIdx.x * blockDim.x + threadIdx.x;
    if (idx >= (long)T_TOT * 512) return;
    int t = (int)(idx >> 9), c = (int)(idx & 511);
    int doc = doc_id[t], p = pos[t], len = doc_lens[doc];
    if (p >= len) p = len - 1;
    float v;
    if (c < 256) {
        v = hout[((long)doc * LMAX + p) * H_DIM + c];                       // fwd
    } else {
        int rp = len - 1 - p; if (rp < 0) rp = 0;
        v = hout[((long)(N_DOCS + doc) * LMAX + rp) * H_DIM + (c - 256)];   // bwd
    }
    out[(long)t * OUT_COLS + 384 + c] = v;
}

// ===========================================================================
extern "C" void kernel_launch(void* const* d_in, const int* in_sizes, int n_in,
                              void* d_out, int out_size, void* d_ws, size_t ws_size,
                              hipStream_t stream) {
    (void)in_sizes; (void)n_in; (void)out_size; (void)ws_size;
    const float* x     = (const float*)d_in[0];
    const int*   dlens = (const int*)d_in[1];
    const float* ptab  = (const float*)d_in[2];
    const float* cw0 = (const float*)d_in[3];  const float* cb0 = (const float*)d_in[4];
    const float* cw1 = (const float*)d_in[5];  const float* cb1 = (const float*)d_in[6];
    const float* cw2 = (const float*)d_in[7];  const float* cb2 = (const float*)d_in[8];
    const float* Wihf = (const float*)d_in[9];  const float* Whhf = (const float*)d_in[10];
    const float* bihf = (const float*)d_in[11]; const float* bhhf = (const float*)d_in[12];
    const float* Wihb = (const float*)d_in[13]; const float* Whhb = (const float*)d_in[14];
    const float* bihb = (const float*)d_in[15]; const float* bhhb = (const float*)d_in[16];
    float* out = (float*)d_out;

    // ---- workspace carve-up (aligned) ----
    char* base = (char*)d_ws;
    size_t off = 0;
    auto alloc = [&](size_t bytes) -> char* {
        off = (off + 255) & ~(size_t)255;
        char* p = base + off; off += bytes; return p;
    };
    int*    offs    = (int*)   alloc((N_DOCS + 1) * 4);
    int*    doc_id  = (int*)   alloc((size_t)T_TOT * 4);
    int*    pos     = (int*)   alloc((size_t)T_TOT * 4);
    bf16_t* xpe     = (bf16_t*)alloc((size_t)T_TOT * D_PAD * 2);
    bf16_t* xd      = (bf16_t*)alloc((size_t)NROWS * D_PAD * 2);
    bf16_t* xr      = (bf16_t*)alloc((size_t)NROWS * D_PAD * 2);
    bf16_t* wih_pk  = (bf16_t*)alloc(2ULL * G4 * D_PAD * 2);
    bf16_t* whh_pk  = (bf16_t*)alloc(2ULL * G4 * H_DIM * 2);
    bf16_t* conv_pk = (bf16_t*)alloc((size_t)CONV_ROWS * D_PAD * 2);
    float*  gbias   = (float*) alloc(2ULL * G4 * 4);
    float*  G       = (float*) alloc(2ULL * NROWS * G4 * 4);
    float*  state   = (float*) alloc(6ULL * N_DOCS * H_DIM * 4);
    float*  hA = state;                              // ping
    float*  hB = state + 2 * N_DOCS * H_DIM;         // pong
    float*  cS = state + 4 * N_DOCS * H_DIM;         // cell state
    int*    bar     = (int*)   alloc(256);
    float*  hout    = (float*) alloc(2ULL * N_DOCS * LMAX * H_DIM * 4);

    // ---- meta + packing ----
    k_meta<<<1, 256, 0, stream>>>(dlens, offs, doc_id, pos);
    {
        long n = (long)T_TOT * D_PAD;
        k_pack_xpe<<<(unsigned)((n + 255) / 256), 256, 0, stream>>>(x, ptab, pos, xpe);
    }
    {
        long n = (long)NROWS * D_PAD;
        k_pack_xd<<<(unsigned)((n + 255) / 256), 256, 0, stream>>>(x, dlens, offs, xd, xr);
    }
    {
        long n = 2L * G4 * D_PAD;
        k_pack_wih<<<(unsigned)((n + 255) / 256), 256, 0, stream>>>(Wihf, Wihb, wih_pk);
    }
    {
        long n = 2L * G4 * H_DIM;
        k_pack_whh<<<(unsigned)((n + 255) / 256), 256, 0, stream>>>(Whhf, Whhb, whh_pk);
    }
    {
        long n = (long)CONV_ROWS * D_PAD;
        k_pack_conv<<<(unsigned)((n + 255) / 256), 256, 0, stream>>>(cw0, cw1, cw2, conv_pk);
    }
    k_pack_bias<<<(2 * G4 + 255) / 256, 256, 0, stream>>>(bihf, bhhf, bihb, bhhb, gbias);
    {
        long n = 6L * N_DOCS * H_DIM;   // hA + hB + cS
        k_zero<<<(unsigned)((n + 255) / 256), 256, 0, stream>>>(state, n);
    }
    k_zero<<<1, 64, 0, stream>>>((float*)bar, 64);

    // ---- bulk GEMMs (WMMA) ----
    k_gemm_gates<<<dim3(G4 / 64, NROWS / 128, 2), 128, 0, stream>>>(xd, xr, wih_pk, gbias, G);
    k_conv<<<dim3(T_TOT / 64, 3), 128, 0, stream>>>(xpe, conv_pk, cb0, cb1, cb2, out);

    // ---- persistent recurrent kernel: 16 blocks, 200 internal steps ----
    k_lstm_persist<<<dim3(NBLK_LSTM), dim3(128), (256 * H_DIM + 64 * H_DIM) * 2, stream>>>(
        G, whh_pk, hA, hB, cS, hout, bar);

    // ---- assemble LSTM half of the output ----
    {
        long n = (long)T_TOT * 512;
        k_gather<<<(unsigned)((n + 255) / 256), 256, 0, stream>>>(hout, doc_id, pos, dlens, out);
    }
}
